// TrajectoryModel4_48507360641635
// MI455X (gfx1250) — matrix operand order repaired and checked by reading, not compiled, not run on hardware
//
#include <hip/hip_runtime.h>
#include <hip/hip_bf16.h>

// ---------------------------------------------------------------------------
// CDNA5 (gfx1250) WMMA types / helpers
// ---------------------------------------------------------------------------
typedef __attribute__((ext_vector_type(16))) __bf16 v16bf;
typedef __attribute__((ext_vector_type(8)))  float  v8f;

union FragBF {
  v16bf v;
  uint4 u[2];
};

union PackBF8 {
  __bf16 h[8];
  uint4  u;
};

__device__ __forceinline__ v8f wmma_bf16(v16bf a, v16bf b, v8f c) {
  // D = A(16x32 bf16) x B(32x16 bf16) + C(16x16 f32)
  return __builtin_amdgcn_wmma_f32_16x16x32_bf16(
      /*neg_a=*/false, a, /*neg_b=*/false, b,
      /*c_mod=*/(short)0, c, /*reuse_a=*/false, /*reuse_b=*/false);
}

// ---------------------------------------------------------------------------
// Weight transpose with zero padding: D[Npad, Kp] = S[K, N]^T (pad with 0)
// ---------------------------------------------------------------------------
__global__ void transpose_pad_kernel(const float* __restrict__ S, float* __restrict__ D,
                                     int K, int N, int Kp, int Npad)
{
  int i = blockIdx.x * 256 + threadIdx.x;
  if (i >= Npad * Kp) return;
  int n = i / Kp, k = i - n * Kp;
  D[i] = (n < N && k < K) ? S[(size_t)k * N + n] : 0.f;
}

// Row padding: D[M, Kp] = S[M, K] with zero-padded tail columns
__global__ void pad_rows_kernel(const float* __restrict__ S, float* __restrict__ D,
                                int M, int K, int Kp)
{
  int i = blockIdx.x * 256 + threadIdx.x;
  if (i >= M * Kp) return;
  int m = i / Kp, c = i - m * Kp;
  D[i] = (c < K) ? S[(size_t)m * K + c] : 0.f;
}

// ---------------------------------------------------------------------------
// GEMM:  C[M,N] = act(A[M,Kp] @ Wt[Npad,Kp]^T + bias + resid)
// Preconditions (guaranteed by the launcher):
//   M % 64 == 0; Kp % 32 == 0; Wt has ceil64(N) zero-padded rows of length Kp;
//   A rows have stride Kp (zero-padded K tail).  ->  staging is branch-free.
// Block: 256 threads = 8 waves; tile 64(M) x 64(N); K steps of 32.
// Wave w: rows 16*(w>>1)..+15, cols 32*(w&1)..+31 (two 16x16 WMMA, shared A).
// ---------------------------------------------------------------------------
__global__ __launch_bounds__(256) void gemm_kernel(
    const float* __restrict__ A, const float* __restrict__ Wt,
    const float* __restrict__ bias, const float* __restrict__ resid,
    float* __restrict__ C, int Kp, int N, int relu)
{
  __shared__ __align__(16) __bf16 sA[64 * 32];   // [m][k]
  __shared__ __align__(16) __bf16 sB[64 * 32];   // [n][k]
  const int tid  = threadIdx.x;
  const int wave = tid >> 5, lane = tid & 31;
  const int half = lane >> 4, l15 = lane & 15;
  const int m0 = blockIdx.x * 64, n0 = blockIdx.y * 64;
  const int wr = wave >> 1, wc = wave & 1;
  // staging coords: 4 threads per row, 8 consecutive k each
  const int sr = tid >> 2, sc = (tid & 3) * 8;

  const float* pA = A  + (size_t)(m0 + sr) * Kp + sc;
  const float* pB = Wt + (size_t)(n0 + sr) * Kp + sc;

  v8f acc0 = {0.f, 0.f, 0.f, 0.f, 0.f, 0.f, 0.f, 0.f};
  v8f acc1 = {0.f, 0.f, 0.f, 0.f, 0.f, 0.f, 0.f, 0.f};

  const int ksteps = Kp >> 5;
  for (int ks = 0; ks < ksteps; ++ks) {
    // ---- stage A tile (64x32): 2x b128 load -> bf16 pack -> 1x ds b128 ----
    {
      const float4 p0 = *(const float4*)(const void*)(pA);
      const float4 p1 = *(const float4*)(const void*)(pA + 4);
      PackBF8 pk;
      pk.h[0] = (__bf16)p0.x; pk.h[1] = (__bf16)p0.y; pk.h[2] = (__bf16)p0.z; pk.h[3] = (__bf16)p0.w;
      pk.h[4] = (__bf16)p1.x; pk.h[5] = (__bf16)p1.y; pk.h[6] = (__bf16)p1.z; pk.h[7] = (__bf16)p1.w;
      *(uint4*)(void*)&sA[sr * 32 + sc] = pk.u;
    }
    // ---- stage B tile (64n x 32k) from padded transposed weights ----
    {
      const float4 p0 = *(const float4*)(const void*)(pB);
      const float4 p1 = *(const float4*)(const void*)(pB + 4);
      PackBF8 pk;
      pk.h[0] = (__bf16)p0.x; pk.h[1] = (__bf16)p0.y; pk.h[2] = (__bf16)p0.z; pk.h[3] = (__bf16)p0.w;
      pk.h[4] = (__bf16)p1.x; pk.h[5] = (__bf16)p1.y; pk.h[6] = (__bf16)p1.z; pk.h[7] = (__bf16)p1.w;
      *(uint4*)(void*)&sB[sr * 32 + sc] = pk.u;
    }
    // prefetch next k-tile while this one is consumed
    if (ks + 1 < ksteps) {
      __builtin_prefetch(pA + 32, 0, 1);
      __builtin_prefetch(pB + 32, 0, 1);
    }
    pA += 32;
    pB += 32;
    __syncthreads();

    FragBF af, b0, b1;
    const __bf16* ap = &sA[(wr * 16 + l15) * 32];
    af.u[0] = *(const uint4*)(const void*)(ap + half * 8);
    af.u[1] = *(const uint4*)(const void*)(ap + 16 + half * 8);
    const __bf16* bp0 = &sB[(wc * 32 + l15) * 32];
    b0.u[0] = *(const uint4*)(const void*)(bp0 + half * 8);
    b0.u[1] = *(const uint4*)(const void*)(bp0 + 16 + half * 8);
    const __bf16* bp1 = &sB[(wc * 32 + 16 + l15) * 32];
    b1.u[0] = *(const uint4*)(const void*)(bp1 + half * 8);
    b1.u[1] = *(const uint4*)(const void*)(bp1 + 16 + half * 8);
    acc0 = wmma_bf16(af.v, b0.v, acc0);
    acc1 = wmma_bf16(af.v, b1.v, acc1);
    __syncthreads();
  }

  // ---- epilogue: C layout row = j+8*half, col = lane&15 ----
#pragma unroll
  for (int ct = 0; ct < 2; ++ct) {
    const v8f acc = ct ? acc1 : acc0;
    const int gn = n0 + wc * 32 + ct * 16 + l15;
    if (gn < N) {
      const float bvv = bias ? bias[gn] : 0.f;
#pragma unroll
      for (int j = 0; j < 8; ++j) {
        const int gm = m0 + wr * 16 + j + 8 * half;   // M % 64 == 0: in bounds
        float v = acc[j] + bvv;
        if (resid) v += resid[(size_t)gm * N + gn];
        if (relu)  v = v > 0.f ? v : 0.f;
        C[(size_t)gm * N + gn] = v;
      }
    }
  }
}

// ---------------------------------------------------------------------------
// Fused MHA:  O = softmax(scale * Q K^T (+mask)) V     (E=128, H=4, D=32)
// One workgroup (4 waves) per (b, h, 16-query tile). Scores in 32KB LDS.
// ---------------------------------------------------------------------------
__global__ __launch_bounds__(128) void attn_kernel(
    const float* __restrict__ Q, const float* __restrict__ K,
    const float* __restrict__ V, float* __restrict__ O,
    const float* __restrict__ mask, int Lq, int Lk, float scale)
{
  __shared__ float sS[16 * 512];
  const int b = blockIdx.x >> 2, h = blockIdx.x & 3;
  const int q0 = blockIdx.y * 16;
  const int tid = threadIdx.x;
  const int wave = tid >> 5, lane = tid & 31;
  const int half = lane >> 4, l15 = lane & 15;

  const float* Qb = Q + ((size_t)b * Lq) * 128 + h * 32;
  const float* Kb = K + ((size_t)b * Lk) * 128 + h * 32;
  const float* Vb = V + ((size_t)b * Lk) * 128 + h * 32;

  // Q fragment (A operand): lane holds query row q0 + (lane&15).
  FragBF qf;
  {
    const int qrow = q0 + l15;
    if (qrow < Lq) {
      const float* g1 = Qb + (size_t)qrow * 128 + half * 8;
      const float4 a0 = *(const float4*)(const void*)(g1);
      const float4 a1 = *(const float4*)(const void*)(g1 + 4);
      const float4 c0 = *(const float4*)(const void*)(g1 + 16);
      const float4 c1 = *(const float4*)(const void*)(g1 + 20);
      qf.v[0] = (__bf16)a0.x; qf.v[1] = (__bf16)a0.y; qf.v[2] = (__bf16)a0.z; qf.v[3] = (__bf16)a0.w;
      qf.v[4] = (__bf16)a1.x; qf.v[5] = (__bf16)a1.y; qf.v[6] = (__bf16)a1.z; qf.v[7] = (__bf16)a1.w;
      qf.v[8]  = (__bf16)c0.x; qf.v[9]  = (__bf16)c0.y; qf.v[10] = (__bf16)c0.z; qf.v[11] = (__bf16)c0.w;
      qf.v[12] = (__bf16)c1.x; qf.v[13] = (__bf16)c1.y; qf.v[14] = (__bf16)c1.z; qf.v[15] = (__bf16)c1.w;
    } else {
#pragma unroll
      for (int e = 0; e < 16; ++e) qf.v[e] = (__bf16)0.f;
    }
  }

  // Phase 1: score tiles. B operand = K^T == natural [key][dim] order.
  const int ntiles = Lk >> 4;
  for (int kt = wave; kt < ntiles; kt += 4) {
    const int key = kt * 16 + l15;
    FragBF kf;
    {
      const float* kp = Kb + (size_t)key * 128 + half * 8;
      const float4 a0 = *(const float4*)(const void*)(kp);
      const float4 a1 = *(const float4*)(const void*)(kp + 4);
      const float4 a2 = *(const float4*)(const void*)(kp + 16);
      const float4 a3 = *(const float4*)(const void*)(kp + 20);
      kf.v[0]  = (__bf16)a0.x; kf.v[1]  = (__bf16)a0.y; kf.v[2]  = (__bf16)a0.z; kf.v[3]  = (__bf16)a0.w;
      kf.v[4]  = (__bf16)a1.x; kf.v[5]  = (__bf16)a1.y; kf.v[6]  = (__bf16)a1.z; kf.v[7]  = (__bf16)a1.w;
      kf.v[8]  = (__bf16)a2.x; kf.v[9]  = (__bf16)a2.y; kf.v[10] = (__bf16)a2.z; kf.v[11] = (__bf16)a2.w;
      kf.v[12] = (__bf16)a3.x; kf.v[13] = (__bf16)a3.y; kf.v[14] = (__bf16)a3.z; kf.v[15] = (__bf16)a3.w;
    }
    v8f acc = {0.f, 0.f, 0.f, 0.f, 0.f, 0.f, 0.f, 0.f};
    acc = wmma_bf16(qf.v, kf.v, acc);
    const bool masked = (mask != nullptr) && (mask[(size_t)b * 4096 + key] <= 0.f);
#pragma unroll
    for (int j = 0; j < 8; ++j) {
      float s = acc[j] * scale;
      if (masked) s = -1e9f;
      sS[(j + 8 * half) * 512 + key] = s;
    }
  }
  __syncthreads();

  // Phase 2: row softmax. 8 lanes per row (16 rows x 8 = 128 threads).
  {
    const int row = wave * 4 + (lane >> 3);
    const int sub = lane & 7;
    float* rp = &sS[row * 512];
    float mx = -1e30f;
    for (int k = sub; k < Lk; k += 8) mx = fmaxf(mx, rp[k]);
#pragma unroll
    for (int off = 4; off > 0; off >>= 1) mx = fmaxf(mx, __shfl_xor(mx, off, 8));
    float sum = 0.f;
    for (int k = sub; k < Lk; k += 8) { float e = __expf(rp[k] - mx); rp[k] = e; sum += e; }
#pragma unroll
    for (int off = 4; off > 0; off >>= 1) sum += __shfl_xor(sum, off, 8);
    const float inv = 1.f / sum;
    for (int k = sub; k < Lk; k += 8) rp[k] *= inv;
  }
  __syncthreads();

  // Phase 3: O = P @ V. Waves 0,1 each produce a 16x16 output tile (D=32).
  if (wave < 2) {
    v8f acc = {0.f, 0.f, 0.f, 0.f, 0.f, 0.f, 0.f, 0.f};
    const int nd = wave * 16 + l15;             // head-local output dim
    for (int kc = 0; kc < Lk; kc += 32) {
      FragBF pf, vf;
      {
        const float* g1 = &sS[l15 * 512 + kc + half * 8];
        const float4 a0 = *(const float4*)(const void*)(g1);
        const float4 a1 = *(const float4*)(const void*)(g1 + 4);
        const float4 c0 = *(const float4*)(const void*)(g1 + 16);
        const float4 c1 = *(const float4*)(const void*)(g1 + 20);
        pf.v[0] = (__bf16)a0.x; pf.v[1] = (__bf16)a0.y; pf.v[2] = (__bf16)a0.z; pf.v[3] = (__bf16)a0.w;
        pf.v[4] = (__bf16)a1.x; pf.v[5] = (__bf16)a1.y; pf.v[6] = (__bf16)a1.z; pf.v[7] = (__bf16)a1.w;
        pf.v[8]  = (__bf16)c0.x; pf.v[9]  = (__bf16)c0.y; pf.v[10] = (__bf16)c0.z; pf.v[11] = (__bf16)c0.w;
        pf.v[12] = (__bf16)c1.x; pf.v[13] = (__bf16)c1.y; pf.v[14] = (__bf16)c1.z; pf.v[15] = (__bf16)c1.w;
      }
      const float* vp = Vb + (size_t)(kc + 8 * half) * 128 + nd;
#pragma unroll
      for (int e = 0; e < 16; ++e) vf.v[e] = (__bf16)vp[(size_t)(e < 8 ? e : e + 8) * 128];
      acc = wmma_bf16(pf.v, vf.v, acc);
    }
    const int gn = h * 32 + nd;
#pragma unroll
    for (int j = 0; j < 8; ++j) {
      int gm = q0 + j + 8 * half;
      if (gm < Lq) O[((size_t)b * Lq + gm) * 128 + gn] = acc[j];
    }
  }
}

// ---------------------------------------------------------------------------
// LayerNorm over E=128: one wave per token, float4 + shfl_xor reductions.
// ---------------------------------------------------------------------------
__global__ __launch_bounds__(256) void ln_kernel(
    const float* __restrict__ X, const float* __restrict__ g,
    const float* __restrict__ bta, float* __restrict__ Y, int M)
{
  const int token = blockIdx.x * 8 + (threadIdx.x >> 5);
  const int lane = threadIdx.x & 31;
  if (token >= M) return;
  const float4 x = *(const float4*)(const void*)(X + (size_t)token * 128 + lane * 4);
  float s = x.x + x.y + x.z + x.w;
#pragma unroll
  for (int off = 16; off > 0; off >>= 1) s += __shfl_xor(s, off, 32);
  const float mu = s * (1.f / 128.f);
  float4 d; d.x = x.x - mu; d.y = x.y - mu; d.z = x.z - mu; d.w = x.w - mu;
  float v = d.x * d.x + d.y * d.y + d.z * d.z + d.w * d.w;
#pragma unroll
  for (int off = 16; off > 0; off >>= 1) v += __shfl_xor(v, off, 32);
  const float inv = rsqrtf(v * (1.f / 128.f) + 1e-5f);
  const float4 gg = *(const float4*)(const void*)(g + lane * 4);
  const float4 bb = *(const float4*)(const void*)(bta + lane * 4);
  float4 y;
  y.x = d.x * inv * gg.x + bb.x;
  y.y = d.y * inv * gg.y + bb.y;
  y.z = d.z * inv * gg.z + bb.z;
  y.w = d.w * inv * gg.w + bb.w;
  *(float4*)(void*)(Y + (size_t)token * 128 + lane * 4) = y;
}

// N=1 classifier head: out[m] = dot(X[m,:], w) + b0 — one wave per token.
__global__ __launch_bounds__(256) void cls_kernel(
    const float* __restrict__ X, const float* __restrict__ w,
    const float* __restrict__ bptr, float* __restrict__ out, int M)
{
  const int token = blockIdx.x * 8 + (threadIdx.x >> 5);
  const int lane = threadIdx.x & 31;
  if (token >= M) return;
  const float4 x  = *(const float4*)(const void*)(X + (size_t)token * 128 + lane * 4);
  const float4 ww = *(const float4*)(const void*)(w + lane * 4);
  float s = x.x * ww.x + x.y * ww.y + x.z * ww.z + x.w * ww.w;
#pragma unroll
  for (int off = 16; off > 0; off >>= 1) s += __shfl_xor(s, off, 32);
  if (lane == 0) out[token] = s + bptr[0];
}

// Build [B*512, 96] zero-padded input: concat(ped_flat[b], modes_flat[k], 0pad)
__global__ void build_inp_kernel(const float* __restrict__ ped,
                                 const float* __restrict__ modes,
                                 float* __restrict__ inp)
{
  size_t i = (size_t)blockIdx.x * 256 + threadIdx.x;
  if (i >= (size_t)65536 * 96) return;
  int c = (int)(i % 96);
  size_t mk = i / 96;
  int k = (int)(mk % 512);
  int b = (int)(mk / 512);
  float v = 0.f;
  if (c < 48)      v = ped[b * 48 + c];
  else if (c < 72) v = modes[k * 24 + (c - 48)];
  inp[i] = v;
}

// Top-100 of 512 per batch row (stable, descending) + gather feature rows.
__global__ __launch_bounds__(256) void topk100_kernel(
    const float* __restrict__ scores, const float* __restrict__ feat,
    float* __restrict__ topf)
{
  __shared__ float sv[512];
  __shared__ float rv[256];
  __shared__ int   ri[256];
  __shared__ int   chosen[100];
  const int b = blockIdx.x, t = threadIdx.x;
  sv[t] = scores[b * 512 + t];
  sv[t + 256] = scores[b * 512 + 256 + t];
  __syncthreads();
  for (int it = 0; it < 100; ++it) {
    float v0 = sv[t], v1 = sv[t + 256];
    float bvv = v0; int bii = t;
    if (v1 > v0) { bvv = v1; bii = t + 256; }   // tie keeps smaller index
    rv[t] = bvv; ri[t] = bii;
    __syncthreads();
    for (int s = 128; s > 0; s >>= 1) {
      if (t < s) {
        float ov = rv[t + s]; int oi = ri[t + s];
        if (ov > rv[t] || (ov == rv[t] && oi < ri[t])) { rv[t] = ov; ri[t] = oi; }
      }
      __syncthreads();
    }
    if (t == 0) { chosen[it] = ri[0]; sv[ri[0]] = -1e30f; }
    __syncthreads();
  }
  for (int e = t; e < 100 * 32; e += 256) {     // float4 gather of 100x128
    int r = e >> 5, c = e & 31;
    *(float4*)(void*)(topf + ((size_t)b * 100 + r) * 128 + c * 4) =
        *(const float4*)(const void*)(feat + ((size_t)b * 512 + chosen[r]) * 128 + c * 4);
  }
}

// softmax over 100, emit scores_nei, top-20 select + gather.
__global__ __launch_bounds__(128) void softmax_topk20_kernel(
    const float* __restrict__ raw, const float* __restrict__ y,
    float* __restrict__ scores_out, float* __restrict__ top2)
{
  __shared__ float sv[100];
  __shared__ int   chosen[20];
  const int b = blockIdx.x, t = threadIdx.x;
  if (t < 100) sv[t] = raw[b * 100 + t];
  __syncthreads();
  if (t == 0) {
    float mx = -1e30f;
    for (int i = 0; i < 100; ++i) mx = fmaxf(mx, sv[i]);
    float sum = 0.f;
    for (int i = 0; i < 100; ++i) { float e = __expf(sv[i] - mx); sv[i] = e; sum += e; }
    float inv = 1.f / sum;
    for (int i = 0; i < 100; ++i) sv[i] *= inv;
  }
  __syncthreads();
  if (t < 100) scores_out[b * 100 + t] = sv[t];
  __syncthreads();
  if (t == 0) {
    for (int it = 0; it < 20; ++it) {
      float bvv = -1e30f; int bii = 0;
      for (int i = 0; i < 100; ++i)
        if (sv[i] > bvv) { bvv = sv[i]; bii = i; }  // first occurrence = stable
      chosen[it] = bii; sv[bii] = -1e30f;
    }
  }
  __syncthreads();
  for (int e = t; e < 20 * 32; e += 128) {      // float4 gather of 20x128
    int r = e >> 5, c = e & 31;
    *(float4*)(void*)(top2 + ((size_t)b * 20 + r) * 128 + c * 4) =
        *(const float4*)(const void*)(y + ((size_t)b * 100 + chosen[r]) * 128 + c * 4);
  }
}

// ---------------------------------------------------------------------------
static inline void launch_gemm(const float* A, const float* Wt, const float* bias,
                               const float* resid, float* C, int M, int Kp, int N,
                               int relu, hipStream_t s)
{
  dim3 g(M / 64, (N + 63) / 64);
  gemm_kernel<<<g, 256, 0, s>>>(A, Wt, bias, resid, C, Kp, N, relu);
}

extern "C" void kernel_launch(void* const* d_in, const int* in_sizes, int n_in,
                              void* d_out, int out_size, void* d_ws, size_t ws_size,
                              hipStream_t stream)
{
  (void)in_sizes; (void)n_in; (void)out_size; (void)ws_size;
  const float* ped_obs   = (const float*)d_in[0];   // [128,8,6]
  const float* neis_obs  = (const float*)d_in[1];   // [128,64,8,6] -> [8192,48]
  const float* modes     = (const float*)d_in[2];   // [512,12,2]
  const float* mask      = (const float*)d_in[3];   // [128,64,64]
  const float* emb_w     = (const float*)d_in[5];
  const float* emb_b     = (const float*)d_in[6];
  const float* enc_wq    = (const float*)d_in[7];
  const float* enc_wk    = (const float*)d_in[8];
  const float* enc_wv    = (const float*)d_in[9];
  const float* enc_wo    = (const float*)d_in[10];
  const float* enc_w1    = (const float*)d_in[11];
  const float* enc_b1    = (const float*)d_in[12];
  const float* enc_w2    = (const float*)d_in[13];
  const float* enc_b2    = (const float*)d_in[14];
  const float* enc_g1    = (const float*)d_in[15];
  const float* enc_bb1   = (const float*)d_in[16];
  const float* enc_g2    = (const float*)d_in[17];
  const float* enc_bb2   = (const float*)d_in[18];
  const float* cls_w     = (const float*)d_in[19];
  const float* cls_b     = (const float*)d_in[20];
  const float* cls2_w    = (const float*)d_in[21];
  const float* cls2_b    = (const float*)d_in[22];
  const float* nei_w     = (const float*)d_in[23];
  const float* nei_b     = (const float*)d_in[24];
  const float* dec_wq    = (const float*)d_in[25];
  const float* dec_wk    = (const float*)d_in[26];
  const float* dec_wv    = (const float*)d_in[27];
  const float* dec_wo    = (const float*)d_in[28];
  const float* dec_w1    = (const float*)d_in[29];
  const float* dec_b1    = (const float*)d_in[30];
  const float* dec_w2    = (const float*)d_in[31];
  const float* dec_b2    = (const float*)d_in[32];
  const float* dec_g1    = (const float*)d_in[33];
  const float* dec_bb1   = (const float*)d_in[34];
  const float* dec_g2    = (const float*)d_in[35];
  const float* dec_bb2   = (const float*)d_in[36];
  const float* reg_w     = (const float*)d_in[37];
  const float* reg_b     = (const float*)d_in[38];
  float* out = (float*)d_out;

  const int   TOK   = 65536;   // B*K = 128*512
  const float scale = 0.17677669529663687f; // 1/sqrt(32)

  // ---- workspace layout (floats) ----
  float* w = (float*)d_ws;
  size_t off = 0;
  float* inp = w + off; off += (size_t)TOK * 96;    // zero-padded K=72 -> 96
  float* xb  = w + off; off += (size_t)TOK * 128;
  float* tb  = w + off; off += (size_t)TOK * 128;
  float* qb  = w + off; off += (size_t)TOK * 128;
  float* kb  = w + off; off += (size_t)TOK * 128;
  float* vb  = w + off; off += (size_t)TOK * 128;
  float* ab  = w + off; off += (size_t)TOK * 128;
  float* hb  = w + off; off += (size_t)TOK * 256;
  // small buffers reuse the `inp` region once embedding is done
  float* scores1 = inp;                                  // 65536
  float* topf    = inp + 65536;                          // 128*100*128
  float* nei_pad = topf + (size_t)128 * 100 * 128;       // 8192*64 (K=48 -> 64)
  float* nei_emb = nei_pad + (size_t)8192 * 64;          // 128*64*128
  float* scores2 = nei_emb + (size_t)128 * 64 * 128;     // 12800
  float* top2    = scores2 + 12800;                      // 128*20*128

  // ---- pre-transpose all weights into zero-padded Wt[ceil64(N), ceil32(K)] ----
  auto xpose = [&](const float* src, int K, int N) -> float* {
    const int Kp   = (K + 31) & ~31;
    const int Npad = (N + 63) & ~63;
    float* dst = w + off; off += (size_t)Npad * Kp;
    transpose_pad_kernel<<<(Npad * Kp + 255) / 256, 256, 0, stream>>>(src, dst, K, N, Kp, Npad);
    return dst;
  };
  float* wt_emb = xpose(emb_w, 72, 128);
  float* wt_eq[2]; float* wt_ek[2]; float* wt_ev[2]; float* wt_eo[2];
  float* wt_e1[2]; float* wt_e2[2];
  for (int L = 0; L < 2; ++L) {
    wt_eq[L] = xpose(enc_wq + (size_t)L * 128 * 128, 128, 128);
    wt_ek[L] = xpose(enc_wk + (size_t)L * 128 * 128, 128, 128);
    wt_ev[L] = xpose(enc_wv + (size_t)L * 128 * 128, 128, 128);
    wt_eo[L] = xpose(enc_wo + (size_t)L * 128 * 128, 128, 128);
    wt_e1[L] = xpose(enc_w1 + (size_t)L * 128 * 256, 128, 256);
    wt_e2[L] = xpose(enc_w2 + (size_t)L * 256 * 128, 256, 128);
  }
  float* wt_nei = xpose(nei_w, 48, 128);
  float* wt_dq = xpose(dec_wq, 128, 128);
  float* wt_dk = xpose(dec_wk, 128, 128);
  float* wt_dv = xpose(dec_wv, 128, 128);
  float* wt_do = xpose(dec_wo, 128, 128);
  float* wt_d1 = xpose(dec_w1, 128, 256);
  float* wt_d2 = xpose(dec_w2, 256, 128);
  float* wt_rg = xpose(reg_w, 128, 24);   // -> [64 rows][128] zero-padded

  // ---- embedding (Kp = 96) ----
  build_inp_kernel<<<(int)(((size_t)65536 * 96 + 255) / 256), 256, 0, stream>>>(ped_obs, modes, inp);
  launch_gemm(inp, wt_emb, emb_b, nullptr, xb, TOK, 96, 128, 0, stream);

  // ---- encoder: 2 layers of self-attention over K=512 per batch ----
  for (int L = 0; L < 2; ++L) {
    launch_gemm(xb, wt_eq[L], nullptr, nullptr, qb, TOK, 128, 128, 0, stream);
    launch_gemm(xb, wt_ek[L], nullptr, nullptr, kb, TOK, 128, 128, 0, stream);
    launch_gemm(xb, wt_ev[L], nullptr, nullptr, vb, TOK, 128, 128, 0, stream);
    attn_kernel<<<dim3(128 * 4, 512 / 16), 128, 0, stream>>>(
        qb, kb, vb, ab, nullptr, 512, 512, scale);
    launch_gemm(ab, wt_eo[L], nullptr, xb, tb, TOK, 128, 128, 0, stream);
    ln_kernel<<<TOK / 8, 256, 0, stream>>>(tb, enc_g1 + L * 128, enc_bb1 + L * 128, xb, TOK);
    launch_gemm(xb, wt_e1[L], enc_b1 + L * 256, nullptr, hb, TOK, 128, 256, 1, stream);
    launch_gemm(hb, wt_e2[L], enc_b2 + L * 128, xb, tb, TOK, 256, 128, 0, stream);
    ln_kernel<<<TOK / 8, 256, 0, stream>>>(tb, enc_g2 + L * 128, enc_bb2 + L * 128, xb, TOK);
  }

  // ---- mode scores + top-100 gather ----
  cls_kernel<<<TOK / 8, 256, 0, stream>>>(xb, cls_w, cls_b, scores1, TOK);
  topk100_kernel<<<128, 256, 0, stream>>>(scores1, xb, topf);

  // ---- neighbor embedding (pad K=48 -> 64) ----
  pad_rows_kernel<<<(8192 * 64 + 255) / 256, 256, 0, stream>>>(neis_obs, nei_pad, 8192, 48, 64);
  launch_gemm(nei_pad, wt_nei, nei_b, nullptr, nei_emb, 8192, 64, 128, 0, stream);

  // ---- decoder: 1 cross-attention layer (Lq=100, Lk=64) ----
  const int MQ = 128 * 100;   // 12800
  const int MK = 128 * 64;    // 8192
  launch_gemm(topf,    wt_dq, nullptr, nullptr, qb, MQ, 128, 128, 0, stream);
  launch_gemm(nei_emb, wt_dk, nullptr, nullptr, kb, MK, 128, 128, 0, stream);
  launch_gemm(nei_emb, wt_dv, nullptr, nullptr, vb, MK, 128, 128, 0, stream);
  attn_kernel<<<dim3(128 * 4, (100 + 15) / 16), 128, 0, stream>>>(
      qb, kb, vb, ab, mask, 100, 64, scale);
  launch_gemm(ab, wt_do, nullptr, topf, tb, MQ, 128, 128, 0, stream);
  ln_kernel<<<MQ / 8, 256, 0, stream>>>(tb, dec_g1, dec_bb1, topf, MQ);
  launch_gemm(topf, wt_d1, dec_b1, nullptr, hb, MQ, 128, 256, 1, stream);
  launch_gemm(hb, wt_d2, dec_b2, topf, tb, MQ, 256, 128, 0, stream);
  ln_kernel<<<MQ / 8, 256, 0, stream>>>(tb, dec_g2, dec_bb2, topf, MQ);

  // ---- neighbor scores: softmax + top-20 + gather ----
  cls_kernel<<<MQ / 8, 256, 0, stream>>>(topf, cls2_w, cls2_b, scores2, MQ);
  // output layout: pred_trajs [128,20,24] then scores_nei [128,100]
  float* scores_nei_out = out + (size_t)128 * 20 * 24;
  softmax_topk20_kernel<<<128, 128, 0, stream>>>(scores2, topf, scores_nei_out, top2);

  // ---- regression head: [2560,128] @ [128,24] + b ----
  launch_gemm(top2, wt_rg, reg_b, nullptr, out, 128 * 20, 128, 24, 0, stream);
}
